// SelfAttention_6648609374945
// MI455X (gfx1250) — matrix operand order repaired and checked by reading, not compiled, not run on hardware
//
#include <hip/hip_runtime.h>
#include <hip/hip_bf16.h>

// Problem constants
#define BATCH 32
#define CCH   256
#define HWP   4096      // 64*64 pixels per plane; H=G=W=64

typedef __attribute__((ext_vector_type(16))) __bf16        v16bf;
typedef __attribute__((ext_vector_type(8)))  float         v8f;
typedef __attribute__((ext_vector_type(8)))  unsigned int  v8u;
typedef __attribute__((ext_vector_type(4)))  int           v4i;

#define AS1 __attribute__((address_space(1)))
#define AS3 __attribute__((address_space(3)))

#if __has_builtin(__builtin_amdgcn_global_load_async_to_lds_b128) && \
    __has_builtin(__builtin_amdgcn_s_wait_asynccnt)
#define HAVE_ASYNC_LDS 1
#else
#define HAVE_ASYNC_LDS 0
#endif

static __device__ __forceinline__ unsigned short f32_to_bf16_rne(float f) {
    unsigned int u = __float_as_uint(f);
    u += 0x7FFFu + ((u >> 16) & 1u);
    return (unsigned short)(u >> 16);
}

// 16-byte global -> LDS copy; async (ASYNCcnt) when the toolchain exposes it.
static __device__ __forceinline__ void g2l_b128(const unsigned short* g, unsigned short* l) {
#if HAVE_ASYNC_LDS
    __builtin_amdgcn_global_load_async_to_lds_b128(
        (AS1 v4i*)(unsigned short*)g, (AS3 v4i*)l, 0, 0);
#else
    *(uint4*)l = *(const uint4*)g;
#endif
}
static __device__ __forceinline__ void g2l_fence() {
#if HAVE_ASYNC_LDS
    __builtin_amdgcn_s_wait_asynccnt(0);
#endif
}

// ---- WMMA fragment loaders (wave32, CDNA5 16x16x32 bf16 layouts) ----
static __device__ __forceinline__ v16bf load_frag_a(const unsigned short* As, int lda,
                                                    int m0, int k0, int lane) {
    int m = m0 + (lane & 15);
    int half = lane >> 4;
    const unsigned short* p = As + m * lda + k0 + 8 * half;
    uint4 q0 = *(const uint4*)(p);        // K slots 0..7
    uint4 q1 = *(const uint4*)(p + 16);   // K slots 8..15 (K += 16)
    v8u r;
    r[0] = q0.x; r[1] = q0.y; r[2] = q0.z; r[3] = q0.w;
    r[4] = q1.x; r[5] = q1.y; r[6] = q1.z; r[7] = q1.w;
    return __builtin_bit_cast(v16bf, r);
}
static __device__ __forceinline__ v16bf load_frag_bt(const unsigned short* Bt, int ldb,
                                                     int n0, int k0, int lane) {
    int n = n0 + (lane & 15);
    int kh = lane >> 4;
    const unsigned short* p = Bt + n * ldb + k0 + 8 * kh;
    uint4 q0 = *(const uint4*)(p);
    uint4 q1 = *(const uint4*)(p + 16);
    v8u r;
    r[0] = q0.x; r[1] = q0.y; r[2] = q0.z; r[3] = q0.w;
    r[4] = q1.x; r[5] = q1.y; r[6] = q1.z; r[7] = q1.w;
    return __builtin_bit_cast(v16bf, r);
}

// =====================================================================
// Flat f32 -> bf16 conversion (weights).
// =====================================================================
__global__ __launch_bounds__(256) void f32_to_bf16_kernel(
    const float* __restrict__ src, unsigned short* __restrict__ dst, int n4) {
    int i = blockIdx.x * 256 + threadIdx.x;
    if (i < n4) {
        float4 v = *(const float4*)(src + i * 4);
        ushort4 o;
        o.x = f32_to_bf16_rne(v.x); o.y = f32_to_bf16_rne(v.y);
        o.z = f32_to_bf16_rne(v.z); o.w = f32_to_bf16_rne(v.w);
        *(ushort4*)(dst + i * 4) = o;
    }
}

// =====================================================================
// x[b,c,p] (f32) -> xt[b,p,c] (bf16): convert + transpose via LDS tiles.
// Grid: (HWP/64, CCH/64, BATCH), 256 threads.
// =====================================================================
__global__ __launch_bounds__(256) void xpose_bf16_kernel(
    const float* __restrict__ x, unsigned short* __restrict__ xt) {
    __shared__ unsigned short t[64 * 72];
    const int tid = threadIdx.x;
    const int p0 = blockIdx.x * 64, c0 = blockIdx.y * 64, b = blockIdx.z;
#pragma unroll
    for (int it = 0; it < 4; ++it) {
        int flat = (tid + it * 256) * 4;
        int c = flat >> 6, p = flat & 63;
        float4 v = *(const float4*)(x + (b * CCH + c0 + c) * HWP + p0 + p);
        ushort4 o;
        o.x = f32_to_bf16_rne(v.x); o.y = f32_to_bf16_rne(v.y);
        o.z = f32_to_bf16_rne(v.z); o.w = f32_to_bf16_rne(v.w);
        *(ushort4*)(t + c * 72 + p) = o;
    }
    __syncthreads();
#pragma unroll
    for (int it = 0; it < 4; ++it) {
        int flat = (tid + it * 256) * 4;
        int p = flat >> 6, c = flat & 63;
        ushort4 o;
        o.x = t[(c + 0) * 72 + p];
        o.y = t[(c + 1) * 72 + p];
        o.z = t[(c + 2) * 72 + p];
        o.w = t[(c + 3) * 72 + p];
        *(ushort4*)(xt + (b * HWP + p0 + p) * CCH + c0 + c) = o;
    }
}

// =====================================================================
// 1x1 conv as GEMM: Out[b,m,p] = sum_c W[m,c]*In[b,c,p] (+bias,+resid)
// W is bf16. B_NMAJOR: input already [b][p][c] (xbT) -> straight b128
// copies (async). Else input is [b][c][p] bf16 (z2) -> transpose scatter.
// Block 256 thr (8 waves); tile M=128 x N=64, K chunked by 64.
// =====================================================================
template <bool B_NMAJOR, bool OUT_RESID>
__global__ __launch_bounds__(256) void conv1x1_wmma(
    const unsigned short* __restrict__ Wb, const float* __restrict__ bias,
    const unsigned short* __restrict__ Xin, const float* __restrict__ resid,
    void* __restrict__ Out) {
    __shared__ unsigned short As[128 * 80];   // [m][k], 160B rows
    __shared__ unsigned short Bt[64 * 80];    // [n=p][k=c]

    const int tid  = threadIdx.x;
    const int lane = tid & 31;
    const int wave = tid >> 5;
    const int p0    = blockIdx.x * 64;
    const int b     = blockIdx.y;
    const int mBase = blockIdx.z * 128;
    const int mW = (wave & 3) * 32;
    const int nW = (wave >> 2) * 32;

    v8f acc[2][2];
#pragma unroll
    for (int i = 0; i < 2; ++i)
#pragma unroll
        for (int j = 0; j < 2; ++j)
#pragma unroll
            for (int r = 0; r < 8; ++r) acc[i][j][r] = 0.f;

    for (int k0 = 0; k0 < CCH; k0 += 64) {
        __syncthreads();
        // A tile: 128x64 bf16, straight 16B copies
#pragma unroll
        for (int it = 0; it < 4; ++it) {
            int flat = (tid + it * 256) * 8;
            int row = flat >> 6, col = flat & 63;
            g2l_b128(Wb + (mBase + row) * CCH + k0 + col, As + row * 80 + col);
        }
        if (B_NMAJOR) {
            // xbT rows are [p][c]: straight 16B copies
#pragma unroll
            for (int it = 0; it < 2; ++it) {
                int flat = (tid + it * 256) * 8;
                int p = flat >> 6, c = flat & 63;
                g2l_b128(Xin + (b * HWP + p0 + p) * CCH + k0 + c, Bt + p * 80 + c);
            }
        } else {
            // [k=c][p] -> Bt[p][c] transpose scatter
#pragma unroll
            for (int it = 0; it < 4; ++it) {
                int flat = (tid + it * 256) * 4;
                int kr = flat >> 6, pc = flat & 63;
                ushort4 v = *(const ushort4*)(Xin + (b * CCH + k0 + kr) * HWP + p0 + pc);
                Bt[(pc + 0) * 80 + kr] = v.x;
                Bt[(pc + 1) * 80 + kr] = v.y;
                Bt[(pc + 2) * 80 + kr] = v.z;
                Bt[(pc + 3) * 80 + kr] = v.w;
            }
        }
        if (k0 + 64 < CCH) {  // prefetch next weight chunk
            __builtin_prefetch(Wb + (mBase + tid) * CCH + k0 + 64 + (tid & 31), 0, 1);
        }
        g2l_fence();
        __syncthreads();
#pragma unroll
        for (int kk = 0; kk < 64; kk += 32) {
            v16bf a0 = load_frag_a(As, 80, mW + 0,  kk, lane);
            v16bf a1 = load_frag_a(As, 80, mW + 16, kk, lane);
            v16bf b0 = load_frag_bt(Bt, 80, nW + 0,  kk, lane);
            v16bf b1 = load_frag_bt(Bt, 80, nW + 16, kk, lane);
            acc[0][0] = __builtin_amdgcn_wmma_f32_16x16x32_bf16(false, a0, false, b0, (short)0, acc[0][0], false, false);
            acc[0][1] = __builtin_amdgcn_wmma_f32_16x16x32_bf16(false, a0, false, b1, (short)0, acc[0][1], false, false);
            acc[1][0] = __builtin_amdgcn_wmma_f32_16x16x32_bf16(false, a1, false, b0, (short)0, acc[1][0], false, false);
            acc[1][1] = __builtin_amdgcn_wmma_f32_16x16x32_bf16(false, a1, false, b1, (short)0, acc[1][1], false, false);
        }
    }

    float*          outf = (float*)Out;
    unsigned short* outh = (unsigned short*)Out;
#pragma unroll
    for (int i = 0; i < 2; ++i)
#pragma unroll
        for (int j = 0; j < 2; ++j) {
            int n = p0 + nW + 16 * j + (lane & 15);
#pragma unroll
            for (int r = 0; r < 8; ++r) {
                int m = mBase + mW + 16 * i + r + 8 * (lane >> 4);
                int idx = (b * CCH + m) * HWP + n;
                float v = acc[i][j][r] + bias[m];
                if (OUT_RESID) outf[idx] = v + resid[idx];
                else           outh[idx] = f32_to_bf16_rne(v);
            }
        }
}

// =====================================================================
// Batched 64x64x64 GEMM, one (b,c) tile per block (8192 blocks, 4 waves).
// =====================================================================
template <bool B_NMAJOR, bool OUT_BF16>
__global__ __launch_bounds__(128) void bgemm64_wmma(
    const unsigned short* __restrict__ Ag, const unsigned short* __restrict__ Bg,
    void* __restrict__ Out) {
    __shared__ unsigned short As[64 * 80];
    __shared__ unsigned short Bt[64 * 80];
    const int tid  = threadIdx.x;
    const int lane = tid & 31;
    const int wave = tid >> 5;
    const int base = blockIdx.x * HWP;

#pragma unroll
    for (int it = 0; it < 4; ++it) {
        int flat = (tid + it * 128) * 8;
        int row = flat >> 6, col = flat & 63;
        g2l_b128(Ag + base + row * 64 + col, As + row * 80 + col);
    }
    if (B_NMAJOR) {
#pragma unroll
        for (int it = 0; it < 4; ++it) {
            int flat = (tid + it * 128) * 8;
            int row = flat >> 6, col = flat & 63;
            g2l_b128(Bg + base + row * 64 + col, Bt + row * 80 + col);
        }
    } else {
#pragma unroll
        for (int it = 0; it < 8; ++it) {
            int flat = (tid + it * 128) * 4;
            int row = flat >> 6, col = flat & 63;
            ushort4 v = *(const ushort4*)(Bg + base + row * 64 + col);
            Bt[(col + 0) * 80 + row] = v.x;
            Bt[(col + 1) * 80 + row] = v.y;
            Bt[(col + 2) * 80 + row] = v.z;
            Bt[(col + 3) * 80 + row] = v.w;
        }
    }
    g2l_fence();
    __syncthreads();

    const int m0 = wave * 16;
    v8f acc[4];
#pragma unroll
    for (int j = 0; j < 4; ++j)
#pragma unroll
        for (int r = 0; r < 8; ++r) acc[j][r] = 0.f;

#pragma unroll
    for (int kk = 0; kk < 64; kk += 32) {
        v16bf a = load_frag_a(As, 80, m0, kk, lane);
#pragma unroll
        for (int j = 0; j < 4; ++j) {
            v16bf bb = load_frag_bt(Bt, 80, j * 16, kk, lane);
            acc[j] = __builtin_amdgcn_wmma_f32_16x16x32_bf16(false, a, false, bb, (short)0, acc[j], false, false);
        }
    }

#pragma unroll
    for (int j = 0; j < 4; ++j) {
        int n = j * 16 + (lane & 15);
#pragma unroll
        for (int r = 0; r < 8; ++r) {
            int m = m0 + r + 8 * (lane >> 4);
            int idx = base + m * 64 + n;
            if (OUT_BF16) ((unsigned short*)Out)[idx] = f32_to_bf16_rne(acc[j][r]);
            else          ((float*)Out)[idx] = acc[j][r];
        }
    }
}

// =====================================================================
// Softmax over channel axis: z[b, :, hg] (stride HWP), 4 hg per thread.
// =====================================================================
__global__ __launch_bounds__(256) void softmax_c_kernel(
    const float* __restrict__ z, unsigned short* __restrict__ zs) {
    int idx = blockIdx.x * 256 + threadIdx.x;
    int b = idx >> 10;                          // HWP/4 = 1024
    int q = (idx & 1023) * 4;
    const float* p = z + b * CCH * HWP + q;

    float mx[4], sm[4];
#pragma unroll
    for (int l = 0; l < 4; ++l) { mx[l] = -3.4e38f; sm[l] = 0.f; }
    for (int c = 0; c < CCH; ++c) {
        float4 v4 = *(const float4*)(p + c * HWP);
        float vv[4] = {v4.x, v4.y, v4.z, v4.w};
#pragma unroll
        for (int l = 0; l < 4; ++l) {
            float nm = fmaxf(mx[l], vv[l]);
            sm[l] = sm[l] * __expf(mx[l] - nm) + __expf(vv[l] - nm);
            mx[l] = nm;
        }
    }
    float inv[4];
#pragma unroll
    for (int l = 0; l < 4; ++l) inv[l] = 1.f / sm[l];

    unsigned short* op = zs + b * CCH * HWP + q;
    for (int c = 0; c < CCH; ++c) {
        float4 v4 = *(const float4*)(p + c * HWP);
        ushort4 o;
        o.x = f32_to_bf16_rne(__expf(v4.x - mx[0]) * inv[0]);
        o.y = f32_to_bf16_rne(__expf(v4.y - mx[1]) * inv[1]);
        o.z = f32_to_bf16_rne(__expf(v4.z - mx[2]) * inv[2]);
        o.w = f32_to_bf16_rne(__expf(v4.w - mx[3]) * inv[3]);
        *(ushort4*)(op + c * HWP) = o;
    }
}

extern "C" void kernel_launch(void* const* d_in, const int* in_sizes, int n_in,
                              void* d_out, int out_size, void* d_ws, size_t ws_size,
                              hipStream_t stream) {
    const float* x  = (const float*)d_in[0];
    const float* wq = (const float*)d_in[1];
    const float* bq = (const float*)d_in[2];
    const float* wv = (const float*)d_in[3];
    const float* bv = (const float*)d_in[4];
    const float* wk = (const float*)d_in[5];
    const float* bk = (const float*)d_in[6];
    const float* wy = (const float*)d_in[7];
    const float* by = (const float*)d_in[8];

    char* ws = (char*)d_ws;
    const size_t BF = 67108864ull;                 // one bf16 [B,C,H,W] tensor
    const size_t WB = 131072ull;                   // one bf16 [256,256] weight
    unsigned short* f1  = (unsigned short*)(ws);
    unsigned short* f2  = (unsigned short*)(ws + BF);
    unsigned short* g1  = (unsigned short*)(ws + 2 * BF);
    unsigned short* xbT = (unsigned short*)(ws + 3 * BF);   // [b][p][c]
    float*          zb  = (float*)(ws + 3 * BF);            // overlaps xbT (dead by stage 2)
    unsigned short* zs  = f1;                               // reuse
    unsigned short* z2  = f2;                               // reuse
    unsigned short* wqb = (unsigned short*)(ws + 5 * BF);
    unsigned short* wvb = (unsigned short*)(ws + 5 * BF + WB);
    unsigned short* wkb = (unsigned short*)(ws + 5 * BF + 2 * WB);
    unsigned short* wyb = (unsigned short*)(ws + 5 * BF + 3 * WB);

    // Pre-convert weights to bf16 (tiny) and x to transposed bf16 [b][p][c]
    int nw4 = CCH * CCH / 4;
    f32_to_bf16_kernel<<<(nw4 + 255) / 256, 256, 0, stream>>>(wq, wqb, nw4);
    f32_to_bf16_kernel<<<(nw4 + 255) / 256, 256, 0, stream>>>(wv, wvb, nw4);
    f32_to_bf16_kernel<<<(nw4 + 255) / 256, 256, 0, stream>>>(wk, wkb, nw4);
    f32_to_bf16_kernel<<<(nw4 + 255) / 256, 256, 0, stream>>>(wy, wyb, nw4);
    xpose_bf16_kernel<<<dim3(HWP / 64, CCH / 64, BATCH), 256, 0, stream>>>(x, xbT);

    dim3 cgrid(HWP / 64, BATCH, CCH / 128);
    // Stage 1: q/v/k projections -> bf16 (x already pixel-major -> pure copies)
    conv1x1_wmma<true, false><<<cgrid, 256, 0, stream>>>(wqb, bq, xbT, nullptr, f1);
    conv1x1_wmma<true, false><<<cgrid, 256, 0, stream>>>(wvb, bv, xbT, nullptr, f2);
    conv1x1_wmma<true, false><<<cgrid, 256, 0, stream>>>(wkb, bk, xbT, nullptr, g1);
    // Stage 2: z[b,c,h,g] = f1 · f2ᵀ
    bgemm64_wmma<true, false><<<BATCH * CCH, 128, 0, stream>>>(f1, f2, zb);
    // Stage 3: softmax over channels -> bf16
    softmax_c_kernel<<<(BATCH * HWP / 4) / 256, 256, 0, stream>>>(zb, zs);
    // Stage 4: z2 = zs · g1
    bgemm64_wmma<false, true><<<BATCH * CCH, 128, 0, stream>>>(zs, g1, z2);
    // Stage 5: output projection + bias + residual -> f32
    conv1x1_wmma<false, true><<<cgrid, 256, 0, stream>>>(wyb, by, z2, x, d_out);
}
